// SwinTransformerBlock_54932631716192
// MI455X (gfx1250) — matrix operand order repaired and checked by reading, not compiled, not run on hardware
//
#include <hip/hip_runtime.h>
#include <hip/hip_bf16.h>
#include <math.h>

typedef _Float16 f16;
typedef __attribute__((ext_vector_type(8)))  _Float16 v8h;
typedef __attribute__((ext_vector_type(16))) _Float16 v16h;
typedef __attribute__((ext_vector_type(8)))  float    v8f;

#define CC     384
#define WS_    7
#define SS_    3
#define NH_    12
#define HD_    32
#define NN_    49
#define NWIN_  64
#define BB_    32
#define BW_    (BB_ * NWIN_)   /* 2048 windows  */
#define TT_    (BW_ * NN_)     /* 100352 tokens == B*H*W */
#define HID_   1536
#define SCALE_ 0.17677669529663689f   /* 32^-0.5 */

__device__ __forceinline__ v8f wmma_f16(v16h a, v16h b, v8f c) {
  return __builtin_amdgcn_wmma_f32_16x16x32_f16(false, a, false, b, (short)0, c,
                                                false, false);
}
__device__ __forceinline__ v16h cat8(v8h lo, v8h hi) {
  return __builtin_shufflevector(lo, hi, 0, 1, 2, 3, 4, 5, 6, 7, 8, 9, 10, 11,
                                 12, 13, 14, 15);
}
// CDNA5 async memory->LDS copy (ASYNCcnt-tracked), 16 bytes per lane.
__device__ __forceinline__ void async_to_lds_b128(void* lds, const void* gaddr) {
  unsigned loff = (unsigned)(unsigned long long)lds;  // flat LDS: addr[31:0]
  asm volatile("global_load_async_to_lds_b128 %0, %1, off"
               :
               : "v"(loff), "v"(gaddr)
               : "memory");
}
__device__ __forceinline__ void wait_asynccnt0() {
  asm volatile("s_wait_asynccnt 0x0" ::: "memory");
}

// ---------------------------------------------------------------------------
// Weight conversion (f32 -> f16) WITH transpose to [N][K] so GEMM B-tiles
// stage with coalesced 16B vector loads.  q-scale folded into qkv w/b.
// ---------------------------------------------------------------------------
__global__ void conv_qkv_w_t(const float* __restrict__ w, f16* __restrict__ wt) {
  int id = blockIdx.x * 256 + threadIdx.x;     // wt[col*384 + k]
  if (id < 3 * CC * CC) {
    int col = id / CC, k = id % CC;
    float s = (col < CC) ? SCALE_ : 1.0f;
    wt[id] = (f16)(w[(size_t)k * (3 * CC) + col] * s);
  }
}
__global__ void conv_qkv_b(const float* __restrict__ b, float* __restrict__ bs) {
  int id = blockIdx.x * 256 + threadIdx.x;
  if (id < 3 * CC) bs[id] = b[id] * ((id < CC) ? SCALE_ : 1.0f);
}
__global__ void conv_t(const float* __restrict__ src, f16* __restrict__ dst,
                       int K, int N) {   // dst[n*K+k] = src[k*N+n]
  int id = blockIdx.x * 256 + threadIdx.x;
  if (id < K * N) {
    int n = id / K, k = id % K;
    dst[id] = (f16)src[(size_t)k * N + n];
  }
}

// ---------------------------------------------------------------------------
// LayerNorm over C=384; optional fused cyclic-shift + window partition gather.
// ---------------------------------------------------------------------------
__global__ __launch_bounds__(128)
void ln_kernel(const float* __restrict__ x, const float* __restrict__ g,
               const float* __restrict__ b, f16* __restrict__ out, int shifted) {
  __shared__ float red[128], red2[128];
  __shared__ float s_mu, s_rs;
  int orow = blockIdx.x;
  int src;
  if (shifted) {
    int bi = orow / 3136, t = orow % 3136;
    int wg = t / NN_, n = t % NN_;
    int hp = (wg >> 3) * WS_ + n / WS_;
    int wp = (wg & 7) * WS_ + n % WS_;
    int hs = hp + SS_; if (hs >= 56) hs -= 56;
    int ws = wp + SS_; if (ws >= 56) ws -= 56;
    src = bi * 3136 + hs * 56 + ws;
  } else {
    src = orow;
  }
  const float* xr = x + (size_t)src * CC;
  int tid = threadIdx.x;
  float v0 = xr[tid], v1 = xr[tid + 128], v2 = xr[tid + 256];
  red[tid]  = v0 + v1 + v2;
  red2[tid] = v0 * v0 + v1 * v1 + v2 * v2;
  __syncthreads();
  for (int o = 64; o > 0; o >>= 1) {
    if (tid < o) { red[tid] += red[tid + o]; red2[tid] += red2[tid + o]; }
    __syncthreads();
  }
  if (tid == 0) {
    float mu  = red[0] / CC;
    float var = red2[0] / CC - mu * mu;
    s_mu = mu;
    s_rs = rsqrtf(var + 1e-5f);
  }
  __syncthreads();
  float mu = s_mu, rs = s_rs;
  f16* op = out + (size_t)orow * CC;
  op[tid]       = (f16)((v0 - mu) * rs * g[tid]       + b[tid]);
  op[tid + 128] = (f16)((v1 - mu) * rs * g[tid + 128] + b[tid + 128]);
  op[tid + 256] = (f16)((v2 - mu) * rs * g[tid + 256] + b[tid + 256]);
}

// ---------------------------------------------------------------------------
// f16 WMMA GEMM, async double-buffered tile staging.
// A:[M,K] row-major f16, Bt:[Nn,K] (pre-transposed weights) f16.
// C tile 64x64 per 128-thread block; k-step 32.
//   mode 0: out16 = f16(val + bias)                       (QKV)
//   mode 1: out32[remap] = resid[remap] + val + bias      (proj + win-reverse)
//   mode 2: out16 = f16(gelu(val + bias))                 (fc1)
//   mode 3: out32 = resid + val + bias                    (fc2 residual)
// ---------------------------------------------------------------------------
__global__ __launch_bounds__(128)
void gemm64(const f16* __restrict__ A, const f16* __restrict__ Bt,
            const float* __restrict__ bias, int M, int K, int Nn, int mode,
            f16* __restrict__ out16, float* __restrict__ out32,
            const float* __restrict__ resid) {
  __shared__ __align__(64) f16 As[2][64][32];
  __shared__ __align__(64) f16 Bs[2][64][32];   // Bs[buf][col][k]
  const int tid = threadIdx.x;
  const int wv = tid >> 5, lane = tid & 31;
  const int lr = lane & 15, hi = lane >> 4;
  const int m0 = blockIdx.x * 64, n0 = blockIdx.y * 64;

  // per-thread staging coordinates: two 16B segments of A, two of B
  const int srow0 = tid >> 2, sseg0 = (tid & 3) * 8;           // seg 0..127
  const int srow1 = (tid + 128) >> 2, sseg1 = ((tid + 128) & 3) * 8;

  auto issue_tile = [&](int buf, int kk) {
    async_to_lds_b128(&As[buf][srow0][sseg0],
                      &A[(size_t)(m0 + srow0) * K + kk + sseg0]);
    async_to_lds_b128(&As[buf][srow1][sseg1],
                      &A[(size_t)(m0 + srow1) * K + kk + sseg1]);
    async_to_lds_b128(&Bs[buf][srow0][sseg0],
                      &Bt[(size_t)(n0 + srow0) * K + kk + sseg0]);
    async_to_lds_b128(&Bs[buf][srow1][sseg1],
                      &Bt[(size_t)(n0 + srow1) * K + kk + sseg1]);
  };

  v8f acc[4] = {};

  int buf = 0;
  issue_tile(0, 0);
  wait_asynccnt0();
  __syncthreads();

  for (int kk = 0; kk < K; kk += 32) {
    if (kk + 32 < K) issue_tile(buf ^ 1, kk + 32);

    // A fragment: row = wv*16 + lr; K = 16*hi + h, contiguous 32B.
    const f16* arow = &As[buf][wv * 16 + lr][0];
    v16h a = *(const v16h*)(arow + 16 * hi);
    for (int j = 0; j < 4; ++j) {
      // B fragment: col = j*16 + lr; K = 16*hi + h, contiguous 32B.
      v16h bf = *(const v16h*)(&Bs[buf][j * 16 + lr][0] + 16 * hi);
      acc[j] = wmma_f16(a, bf, acc[j]);
    }

    wait_asynccnt0();   // next tile landed in LDS
    __syncthreads();    // all waves done reading current tile
    buf ^= 1;
  }

  for (int j = 0; j < 4; ++j) {
    int col = n0 + j * 16 + lr;
    float bv = bias ? bias[col] : 0.0f;
    for (int r = 0; r < 8; ++r) {
      int m = m0 + wv * 16 + r + 8 * hi;
      float v = acc[j][r] + bv;
      if (mode == 0) {
        out16[(size_t)m * Nn + col] = (f16)v;
      } else if (mode == 2) {
        float gl = 0.5f * v * (1.0f + erff(v * 0.70710678118654752f));
        out16[(size_t)m * Nn + col] = (f16)gl;
      } else if (mode == 1) {
        int bi = m / 3136, t = m % 3136;
        int wg = t / NN_, n = t % NN_;
        int hp = (wg >> 3) * WS_ + n / WS_;
        int wp = (wg & 7) * WS_ + n % WS_;
        int hf = hp + SS_; if (hf >= 56) hf -= 56;
        int wf = wp + SS_; if (wf >= 56) wf -= 56;
        size_t orow = (size_t)bi * 3136 + hf * 56 + wf;
        out32[orow * CC + col] = resid[orow * CC + col] + v;
      } else {  // mode 3
        out32[(size_t)m * Nn + col] = resid[(size_t)m * Nn + col] + v;
      }
    }
  }
}

// ---------------------------------------------------------------------------
// Fused window attention: one block per (window, head). N=49 padded to 64,
// HD=32 => QK^T is a single k=32 WMMA chunk per output tile.
// ---------------------------------------------------------------------------
__global__ __launch_bounds__(128)
void attn_kernel(const f16* __restrict__ qkv, const float* __restrict__ bias_table,
                 f16* __restrict__ o16) {
  __shared__ __align__(64) f16 Qs[64][32];
  __shared__ __align__(64) f16 Ks[64][32];
  __shared__ __align__(64) f16 Vt[32][64];   // Vt[d][token]
  __shared__ __align__(64) f16 S16[64][64];
  __shared__ float Sf[64][64];
  __shared__ int cnt[64];

  const int wg = blockIdx.x, head = blockIdx.y;
  const int tid = threadIdx.x;
  const int wv = tid >> 5, lane = tid & 31;
  const int lr = lane & 15, hi = lane >> 4;

  // stage q/k/v for this (window, head); pad rows 49..63 with zero
  for (int i = 0; i < 16; ++i) {
    int e = i * 128 + tid;              // 0..2047
    int row = e >> 5, d = e & 31;
    f16 qv = (f16)0, kv = (f16)0, vv = (f16)0;
    if (row < NN_) {
      size_t base = (size_t)(wg * NN_ + row) * (3 * CC) + head * HD_ + d;
      qv = qkv[base];
      kv = qkv[base + CC];
      vv = qkv[base + 2 * CC];
    }
    Qs[row][d] = qv;
    Ks[row][d] = kv;
    Vt[d][row] = vv;
  }
  if (tid < 64) {
    int n = tid, c = 9;
    if (n < NN_) {
      int wl = wg & 63;
      int hp = (wl >> 3) * WS_ + n / WS_;
      int wp = (wl & 7) * WS_ + n % WS_;
      int rh = hp < 49 ? 0 : (hp < 53 ? 1 : 2);
      int rw = wp < 49 ? 0 : (wp < 53 ? 1 : 2);
      c = rh * 3 + rw;
    }
    cnt[n] = c;
  }
  __syncthreads();

  // S = Q @ K^T   (one k=32 chunk, 4 col tiles per wave)
  {
    v8f sacc[4] = {};
    const f16* qrow = &Qs[wv * 16 + lr][0];
    v16h a = *(const v16h*)(qrow + 16 * hi);
    for (int j = 0; j < 4; ++j) {
      v16h bf = *(const v16h*)(&Ks[j * 16 + lr][0] + 16 * hi);
      sacc[j] = wmma_f16(a, bf, sacc[j]);
    }
    for (int j = 0; j < 4; ++j)
      for (int r = 0; r < 8; ++r)
        Sf[wv * 16 + r + 8 * hi][j * 16 + lr] = sacc[j][r];
  }
  __syncthreads();

  // per-row softmax with analytic relative-position bias + shift mask
  if (tid < 64) {
    int i = tid;
    if (i < NN_) {
      int r1 = i / WS_, c1 = i % WS_, ci = cnt[i];
      float mx = -1e30f;
      for (int j = 0; j < NN_; ++j) {
        int r2 = j / WS_, c2 = j % WS_;
        int idx = (r1 - r2 + WS_ - 1) * (2 * WS_ - 1) + (c1 - c2 + WS_ - 1);
        float s = Sf[i][j] + bias_table[idx * NH_ + head];
        if (ci != cnt[j]) s -= 100.0f;
        Sf[i][j] = s;
        mx = fmaxf(mx, s);
      }
      float sum = 0.0f;
      for (int j = 0; j < NN_; ++j) {
        float e = expf(Sf[i][j] - mx);
        Sf[i][j] = e;
        sum += e;
      }
      float inv = 1.0f / sum;
      for (int j = 0; j < NN_; ++j) S16[i][j] = (f16)(Sf[i][j] * inv);
      for (int j = NN_; j < 64; ++j) S16[i][j] = (f16)0;
    } else {
      for (int j = 0; j < 64; ++j) S16[i][j] = (f16)0;
    }
  }
  __syncthreads();

  // O = S @ V   (2 k-chunks of 32 tokens, 2 col tiles of 16 dims)
  v8f oacc[2] = {};
  for (int kc = 0; kc < 2; ++kc) {
    const f16* srow = &S16[wv * 16 + lr][32 * kc];
    v16h a = *(const v16h*)(srow + 16 * hi);
    for (int j = 0; j < 2; ++j) {
      v16h bf = *(const v16h*)(&Vt[j * 16 + lr][0] + 32 * kc + 16 * hi);
      oacc[j] = wmma_f16(a, bf, oacc[j]);
    }
  }
  for (int j = 0; j < 2; ++j)
    for (int r = 0; r < 8; ++r) {
      int m = wv * 16 + r + 8 * hi;
      if (m < NN_)
        o16[(size_t)(wg * NN_ + m) * CC + head * HD_ + j * 16 + lr] =
            (f16)oacc[j][r];
    }
}

// ---------------------------------------------------------------------------
extern "C" void kernel_launch(void* const* d_in, const int* in_sizes, int n_in,
                              void* d_out, int out_size, void* d_ws, size_t ws_size,
                              hipStream_t stream) {
  (void)in_sizes; (void)n_in; (void)out_size; (void)ws_size;
  const float* x       = (const float*)d_in[0];
  const float* n1g     = (const float*)d_in[1];
  const float* n1b     = (const float*)d_in[2];
  const float* qkv_w   = (const float*)d_in[3];
  const float* qkv_b   = (const float*)d_in[4];
  const float* btab    = (const float*)d_in[5];
  const float* proj_w  = (const float*)d_in[6];
  const float* proj_b  = (const float*)d_in[7];
  const float* n2g     = (const float*)d_in[8];
  const float* n2b     = (const float*)d_in[9];
  const float* fc1_w   = (const float*)d_in[10];
  const float* fc1_b   = (const float*)d_in[11];
  const float* fc2_w   = (const float*)d_in[12];
  const float* fc2_b   = (const float*)d_in[13];
  float* out = (float*)d_out;

  char* ws = (char*)d_ws;
  size_t off = 0;
  auto take = [&](size_t bytes) -> char* {
    char* p = ws + off;
    off = (off + bytes + 255) & ~(size_t)255;
    return p;
  };
  f16*   qkv_wt  = (f16*)take((size_t)3 * CC * CC * 2);   // [1152][384]
  float* qkv_bs  = (float*)take((size_t)3 * CC * 4);
  f16*   proj_wt = (f16*)take((size_t)CC * CC * 2);       // [384][384]
  f16*   fc1_wt  = (f16*)take((size_t)HID_ * CC * 2);     // [1536][384]
  f16*   fc2_wt  = (f16*)take((size_t)CC * HID_ * 2);     // [384][1536]
  f16*   xw16    = (f16*)take((size_t)TT_ * CC * 2);      // also reused as m16
  f16*   qkv16   = (f16*)take((size_t)TT_ * HID_ * 2);    // also reused as h16
  f16*   o16     = (f16*)take((size_t)TT_ * CC * 2);
  f16*   m16 = xw16;
  f16*   h16 = qkv16;

  // weight conversion + transpose (one-time per launch)
  conv_qkv_w_t<<<(3 * CC * CC + 255) / 256, 256, 0, stream>>>(qkv_w, qkv_wt);
  conv_qkv_b<<<(3 * CC + 255) / 256, 256, 0, stream>>>(qkv_b, qkv_bs);
  conv_t<<<(CC * CC + 255) / 256, 256, 0, stream>>>(proj_w, proj_wt, CC, CC);
  conv_t<<<(CC * HID_ + 255) / 256, 256, 0, stream>>>(fc1_w, fc1_wt, CC, HID_);
  conv_t<<<(HID_ * CC + 255) / 256, 256, 0, stream>>>(fc2_w, fc2_wt, HID_, CC);

  // LN1 + cyclic shift + window partition -> xw16 [T, C]
  ln_kernel<<<TT_, 128, 0, stream>>>(x, n1g, n1b, xw16, 1);

  // QKV GEMM: [T,384] x [384,1152] (q-scale folded) -> qkv16
  gemm64<<<dim3(TT_ / 64, (3 * CC) / 64), 128, 0, stream>>>(
      xw16, qkv_wt, qkv_bs, TT_, CC, 3 * CC, 0, qkv16, nullptr, nullptr);

  // fused window attention -> o16 [T, C]
  attn_kernel<<<dim3(BW_, NH_), 128, 0, stream>>>(qkv16, btab, o16);

  // proj GEMM + window reverse + unshift + residual -> d_out (x1)
  gemm64<<<dim3(TT_ / 64, CC / 64), 128, 0, stream>>>(
      o16, proj_wt, proj_b, TT_, CC, CC, 1, nullptr, out, x);

  // LN2 -> m16
  ln_kernel<<<TT_, 128, 0, stream>>>(out, n2g, n2b, m16, 0);

  // fc1 + exact GELU -> h16 [T, 1536]
  gemm64<<<dim3(TT_ / 64, HID_ / 64), 128, 0, stream>>>(
      m16, fc1_wt, fc1_b, TT_, CC, HID_, 2, h16, nullptr, nullptr);

  // fc2 + residual accumulate -> d_out
  gemm64<<<dim3(TT_ / 64, CC / 64), 128, 0, stream>>>(
      h16, fc2_wt, fc2_b, TT_, HID_, CC, 3, nullptr, out, out);
}
